// DenseAttnProcessor_23794118820165
// MI455X (gfx1250) — matrix operand order repaired and checked by reading, not compiled, not run on hardware
//
#include <hip/hip_runtime.h>
#include <hip/hip_bf16.h>

// ---------------------------------------------------------------------------
// DenseAttnProcessor for MI455X (gfx1250, wave32, WMMA)
//   B=8, HW=4096, C=1024, T=77, CT=2048, HEADS=16, HEAD_DIM=64, INNER=1024
// f32->f16 convert (weights transposed) -> WMMA GEMMs (Q/K/V) ->
// zero-padded K/V repack -> WMMA attention (bias + masked softmax) ->
// WMMA O-proj (+bias+residual).
// GEMM staging: double-buffered GLOBAL_LOAD_ASYNC_TO_LDS_B128 pipeline,
// synchronized with partial s_wait_asynccnt (async loads complete in order).
// ---------------------------------------------------------------------------

typedef _Float16 h16;
typedef __attribute__((ext_vector_type(16))) _Float16 v16h;
typedef __attribute__((ext_vector_type(8)))  float    v8f;

union Frag16 {            // one WMMA 16-bit A/B operand (8 VGPRs)
    v16h          v;
    unsigned int  u[8];
};

#if defined(__has_builtin)
#if __has_builtin(__builtin_amdgcn_global_load_async_to_lds_b128) && \
    __has_builtin(__builtin_amdgcn_s_wait_asynccnt)
#define USE_ASYNC_LDS 1
// builtin signature (from toolchain diagnostics):
//   void (v4i addrspace(1)*, v4i addrspace(3)*, imm int offset, imm int cpol)
typedef int v4i __attribute__((vector_size(16)));
typedef __attribute__((address_space(1))) v4i* as1_v4i;
typedef __attribute__((address_space(3))) v4i* as3_v4i;
#endif
#endif

#define BATCH   8
#define HW      4096
#define CH      1024
#define TT      77
#define TPAD    96
#define CTXC    2048
#define NHEAD   16
#define HDIM    64
#define INNER   1024
#define SUPPRESS 20.0f

#define LDS_STRIDE 40   // halves; 80 B rows -> 16-B aligned, conflict-free

// ---------------------------------------------------------------------------
// fp32 -> fp16 conversion (activations)
// ---------------------------------------------------------------------------
__global__ void cvt_f32_f16(const float* __restrict__ s, h16* __restrict__ d,
                            long long n) {
    long long i = (long long)blockIdx.x * blockDim.x + threadIdx.x;
    long long stride = (long long)gridDim.x * blockDim.x;
    for (; i < n; i += stride) d[i] = (h16)s[i];
}

// ---------------------------------------------------------------------------
// fp32 [K][N] -> fp16 transposed [N][K] (weights), LDS-tiled 32x32
// grid (N/32, K/32), block 256 (32x8)
// ---------------------------------------------------------------------------
__global__ __launch_bounds__(256)
void cvt_transpose_f32_f16(const float* __restrict__ W, h16* __restrict__ Wt,
                           int K, int N) {
    __shared__ h16 tile[32][33];
    const int n0 = blockIdx.x * 32, k0 = blockIdx.y * 32;
    const int tx = threadIdx.x & 31, ty = threadIdx.x >> 5;
    #pragma unroll
    for (int i = 0; i < 4; ++i)
        tile[ty + i * 8][tx] = (h16)W[(size_t)(k0 + ty + i * 8) * N + n0 + tx];
    __syncthreads();
    #pragma unroll
    for (int i = 0; i < 4; ++i)
        Wt[(size_t)(n0 + ty + i * 8) * K + k0 + tx] = tile[tx][ty + i * 8];
}

// ---------------------------------------------------------------------------
// Repack compact K/V [B*77, INNER] into zero-padded per-head layouts:
//   Kp[b][h][t=96][d=64]  (row-major over d)   -> stage-1 B fragments
//   Vt[b][h][d=64][t=96]  (transposed)         -> stage-2 B fragments
// ---------------------------------------------------------------------------
__global__ __launch_bounds__(64)
void repack_kv(const h16* __restrict__ Kc, const h16* __restrict__ Vc,
               h16* __restrict__ Kp, h16* __restrict__ Vt) {
    const int d = threadIdx.x;
    const int t = blockIdx.x;
    const int h = blockIdx.y;
    const int b = blockIdx.z;
    const int bh = b * NHEAD + h;

    h16 kv = (h16)0.0f, vv = (h16)0.0f;
    if (t < TT) {
        size_t src = (size_t)(b * TT + t) * INNER + h * HDIM + d;
        kv = Kc[src];
        vv = Vc[src];
    }
    Kp[((size_t)bh * TPAD + t) * HDIM + d] = kv;
    Vt[((size_t)bh * HDIM + d) * TPAD + t] = vv;
}

// ---------------------------------------------------------------------------
// Tiled WMMA GEMM: C[M,N] = A[M,K] * Bt[N,K]^T  (A row-major, Bt = B^T)
// Block = 128 threads = 4 wave32; block tile 128(M) x 64(N); k-step 32.
// Each wave owns a 32x64 sub-tile (2 M-frags x 4 N-frags = 8 WMMA/k-step).
// Async path: double-buffered LDS, next tile issued before waiting, then
// s_wait_asynccnt(6) drains only the current tile's 6 in-order async loads.
// STORE_F16: store f16; else store f32 + bias[col] + residual[row,col].
// GUARD_M : bounds-check M (only the 616-row K/V projections; sync staging).
// ---------------------------------------------------------------------------
template <bool STORE_F16, bool GUARD_M>
__global__ __launch_bounds__(128)
void gemm_wmma(const h16* __restrict__ A, const h16* __restrict__ Bt,
               void* __restrict__ Cout,
               const float* __restrict__ bias,
               const float* __restrict__ residual,
               int M, int N, int K) {
    __shared__ h16 aS[2][128 * LDS_STRIDE];  // 128 rows x 32 k, double-buffered
    __shared__ h16 bS[2][64 * LDS_STRIDE];   // 64 cols  x 32 k, double-buffered

    const int tid  = threadIdx.x;
    const int lane = tid & 31;
    const int wave = tid >> 5;
    const int hs   = lane >> 4;   // half-wave select
    const int ln   = lane & 15;
    const int row0 = blockIdx.x * 128;
    const int col0 = blockIdx.y * 64;

    v8f acc[2][4] = {};

    // 8 WMMAs on the staged tile in LDS buffer p
    auto compute_tile = [&](int p) {
        // A fragments: lane = M-row; k in {0..15} (+16 if hs)
        Frag16 af[2];
        #pragma unroll
        for (int mi = 0; mi < 2; ++mi) {
            const h16* ap = &aS[p][(wave * 32 + mi * 16 + ln) * LDS_STRIDE + hs * 16];
            #pragma unroll
            for (int j = 0; j < 4; ++j)
                af[mi].u[j] = *(const unsigned int*)(ap + 2 * j);            // 16B
            #pragma unroll
            for (int j = 4; j < 8; ++j)
                af[mi].u[j] = *(const unsigned int*)(ap + 8 + 2 * (j - 4)); // 16B
        }
        // 4 N-tiles of 16; B fragment = 32B contiguous (2x b128)
        #pragma unroll
        for (int t = 0; t < 4; ++t) {
            Frag16 bf;
            const h16* bp = &bS[p][(t * 16 + ln) * LDS_STRIDE + hs * 16];
            #pragma unroll
            for (int j = 0; j < 8; ++j)
                bf.u[j] = *(const unsigned int*)(bp + 2 * j);
            #pragma unroll
            for (int mi = 0; mi < 2; ++mi)
                acc[mi][t] = __builtin_amdgcn_wmma_f32_16x16x32_f16(
                    false, af[mi].v, false, bf.v, (short)0, acc[mi][t], false, false);
        }
    };

#if defined(USE_ASYNC_LDS)
    if constexpr (!GUARD_M) {
        // issue one k-tile's 6 async b128 loads per thread into buffer p
        auto issue_tile = [&](int kb, int p) {
            #pragma unroll
            for (int i = 0; i < 4; ++i) {          // A: 128 rows x 4 chunks of 16B
                int d = i * 128 + tid, r = d >> 2, c = d & 3;
                __builtin_amdgcn_global_load_async_to_lds_b128(
                    (as1_v4i)(A + (size_t)(row0 + r) * K + kb + c * 8),
                    (as3_v4i)(&aS[p][r * LDS_STRIDE + c * 8]), 0, 0);
            }
            #pragma unroll
            for (int i = 0; i < 2; ++i) {          // B: 64 rows x 4 chunks of 16B
                int d = i * 128 + tid, r = d >> 2, c = d & 3;
                __builtin_amdgcn_global_load_async_to_lds_b128(
                    (as1_v4i)(Bt + (size_t)(col0 + r) * K + kb + c * 8),
                    (as3_v4i)(&bS[p][r * LDS_STRIDE + c * 8]), 0, 0);
            }
        };
        issue_tile(0, 0);                          // prologue
        for (int kb = 0; kb < K; kb += 32) {
            int p = (kb >> 5) & 1;
            if (kb + 32 < K) {
                issue_tile(kb + 32, p ^ 1);        // prefetch next tile first
                __builtin_amdgcn_s_wait_asynccnt(6); // drain current 6 (in-order)
            } else {
                __builtin_amdgcn_s_wait_asynccnt(0);
            }
            __syncthreads();
            compute_tile(p);
            __syncthreads();                       // readers done before overwrite
        }
    } else
#endif
    {
        for (int kb = 0; kb < K; kb += 32) {
            // synchronous staging into buffer 0 (coalesced dwords, M-guarded)
            #pragma unroll
            for (int i = 0; i < 16; ++i) {
                int d  = i * 128 + tid;
                int ar = d >> 4, ak = (d & 15) << 1;
                unsigned int av;
                if constexpr (GUARD_M) {
                    int gr = row0 + ar;
                    av = (gr < M)
                       ? *(const unsigned int*)(A + (size_t)gr * K + kb + ak) : 0u;
                } else {
                    av = *(const unsigned int*)(A + (size_t)(row0 + ar) * K + kb + ak);
                }
                *(unsigned int*)(&aS[0][ar * LDS_STRIDE + ak]) = av;
            }
            #pragma unroll
            for (int i = 0; i < 8; ++i) {
                int d  = i * 128 + tid;
                int br = d >> 4, bk = (d & 15) << 1;
                *(unsigned int*)(&bS[0][br * LDS_STRIDE + bk]) =
                    *(const unsigned int*)(Bt + (size_t)(col0 + br) * K + kb + bk);
            }
            if (kb + 32 < K)                       // global_prefetch_b8
                __builtin_prefetch(Bt + (size_t)(col0 + (tid & 63)) * K + kb + 32);
            __syncthreads();
            compute_tile(0);
            __syncthreads();
        }
    }

    // ---- epilogue: C/D layout -> row = r + 8*hs, col = ln
    #pragma unroll
    for (int mi = 0; mi < 2; ++mi)
        #pragma unroll
        for (int t = 0; t < 4; ++t)
            #pragma unroll
            for (int r = 0; r < 8; ++r) {
                int row = row0 + wave * 32 + mi * 16 + r + hs * 8;
                int col = col0 + t * 16 + ln;
                if (!GUARD_M || row < M) {
                    size_t idx = (size_t)row * N + col;
                    if constexpr (STORE_F16) {
                        ((h16*)Cout)[idx] = (h16)acc[mi][t][r];
                    } else {
                        ((float*)Cout)[idx] = acc[mi][t][r] + bias[col] + residual[idx];
                    }
                }
            }
}

// ---------------------------------------------------------------------------
// Attention: one wave32 per (b, h, 16-query tile).
//   logits = Q Kp^T * 1/8  (WMMA over padded T=96, unconditional b128 loads)
//   + suppression bias on token columns (set semantics, B overwrites A)
//   masked softmax (shfl_xor reductions inside 16-lane halves)
//   ctx = P Vt (WMMA; probs re-laid-out through LDS; Vt gives dword pairs)
// ---------------------------------------------------------------------------
__global__ __launch_bounds__(32)
void attn_wmma(const h16* __restrict__ Q, const h16* __restrict__ Kp,
               const h16* __restrict__ Vt, const float* __restrict__ mA,
               const float* __restrict__ mB, const int* __restrict__ iA,
               const int* __restrict__ iB, h16* __restrict__ Ctx) {
    __shared__ h16 pS[16 * TPAD];                 // probs tile 16 x 96

    const int lane = threadIdx.x & 31;
    const int hs   = lane >> 4;
    const int ln   = lane & 15;
    const int q0   = blockIdx.x * 16;
    const int h    = blockIdx.y;
    const int b    = blockIdx.z;
    const int bh   = b * NHEAD + h;
    const size_t qrow  = (size_t)b * HW + q0;     // base row in Q / Ctx
    const h16* kpB = Kp + (size_t)bh * TPAD * HDIM;
    const h16* vtB = Vt + (size_t)bh * HDIM * TPAD;

    // ---------------- stage 1: logits = Q Kp^T -----------------------------
    v8f accL[6] = {};
    #pragma unroll
    for (int kc = 0; kc < HDIM; kc += 32) {
        Frag16 af;
        #pragma unroll
        for (int j = 0; j < 8; ++j) {
            int k0 = 2 * j + hs * 16;
            af.u[j] = *(const unsigned int*)(Q + (qrow + ln) * INNER + h * HDIM + kc + k0);
        }
        #pragma unroll
        for (int jt = 0; jt < 6; ++jt) {          // 6 key tiles of 16 -> 96
            Frag16 bf;
            #pragma unroll
            for (int j = 0; j < 8; ++j) {
                int kk  = kc + hs * 16 + 2 * j;   // contraction index (d)
                int key = jt * 16 + ln;           // key column (padded, no guard)
                bf.u[j] = *(const unsigned int*)(kpB + (size_t)key * HDIM + kk);
            }
            accL[jt] = __builtin_amdgcn_wmma_f32_16x16x32_f16(
                false, af.v, false, bf.v, (short)0, accL[jt], false, false);
        }
    }

    // ---------------- bias + masked softmax --------------------------------
    float sA[8], sB[8];
    #pragma unroll
    for (int r = 0; r < 8; ++r) {
        int q = q0 + r + hs * 8;
        sA[r] = -SUPPRESS * (1.0f - mA[q]);
        sB[r] = -SUPPRESS * (1.0f - mB[q]);
    }
    const int ia0 = iA[0], ia1 = iA[1], ib0 = iB[0], ib1 = iB[1];

    float vals[6][8];
    float rmax[8], rsum[8];
    #pragma unroll
    for (int r = 0; r < 8; ++r) rmax[r] = -3.0e38f;

    #pragma unroll
    for (int jt = 0; jt < 6; ++jt) {
        int col = jt * 16 + ln;
        #pragma unroll
        for (int r = 0; r < 8; ++r) {
            float x = accL[jt][r] * 0.125f;       // 1/sqrt(64)
            if (col == ib0 || col == ib1)       x += sB[r];   // B overwrites A
            else if (col == ia0 || col == ia1)  x += sA[r];
            if (col >= TT) x = -1.0e30f;          // mask padded keys
            vals[jt][r] = x;
            rmax[r] = fmaxf(rmax[r], x);
        }
    }
    #pragma unroll
    for (int r = 0; r < 8; ++r) {                 // row lives in one 16-lane half
        float m = rmax[r];
        m = fmaxf(m, __shfl_xor(m, 1));
        m = fmaxf(m, __shfl_xor(m, 2));
        m = fmaxf(m, __shfl_xor(m, 4));
        m = fmaxf(m, __shfl_xor(m, 8));
        rmax[r] = m;
        rsum[r] = 0.0f;
    }
    #pragma unroll
    for (int jt = 0; jt < 6; ++jt)
        #pragma unroll
        for (int r = 0; r < 8; ++r) {
            float e = __expf(vals[jt][r] - rmax[r]);
            vals[jt][r] = e;
            rsum[r] += e;
        }
    #pragma unroll
    for (int r = 0; r < 8; ++r) {
        float s = rsum[r];
        s += __shfl_xor(s, 1);
        s += __shfl_xor(s, 2);
        s += __shfl_xor(s, 4);
        s += __shfl_xor(s, 8);
        rsum[r] = 1.0f / s;
    }
    // write normalized probs to LDS (C-layout -> plain 16x96 matrix)
    #pragma unroll
    for (int jt = 0; jt < 6; ++jt)
        #pragma unroll
        for (int r = 0; r < 8; ++r)
            pS[(r + hs * 8) * TPAD + jt * 16 + ln] = (h16)(vals[jt][r] * rsum[r]);
    __syncthreads();

    // ---------------- stage 2: ctx = P Vt ----------------------------------
    v8f accO[4] = {};
    #pragma unroll
    for (int kc = 0; kc < 3; ++kc) {              // 3 key chunks of 32
        Frag16 af;
        #pragma unroll
        for (int j = 0; j < 8; ++j) {
            int k0 = kc * 32 + 2 * j + hs * 16;
            af.u[j] = *(const unsigned int*)(pS + ln * TPAD + k0);
        }
        #pragma unroll
        for (int jn = 0; jn < 4; ++jn) {          // 4 d tiles of 16 -> 64
            Frag16 bf;
            #pragma unroll
            for (int j = 0; j < 8; ++j) {
                int key  = kc * 32 + hs * 16 + 2 * j;     // even -> aligned dword
                int dcol = jn * 16 + ln;
                bf.u[j] = *(const unsigned int*)(vtB + (size_t)dcol * TPAD + key);
            }
            accO[jn] = __builtin_amdgcn_wmma_f32_16x16x32_f16(
                false, af.v, false, bf.v, (short)0, accO[jn], false, false);
        }
    }
    // store ctx in [B*HW, INNER] layout (== transpose(0,2,1,3).reshape)
    #pragma unroll
    for (int jn = 0; jn < 4; ++jn)
        #pragma unroll
        for (int r = 0; r < 8; ++r)
            Ctx[(qrow + r + hs * 8) * INNER + h * HDIM + jn * 16 + ln] =
                (h16)accO[jn][r];
}

// ---------------------------------------------------------------------------
// launcher
// ---------------------------------------------------------------------------
extern "C" void kernel_launch(void* const* d_in, const int* in_sizes, int n_in,
                              void* d_out, int out_size, void* d_ws, size_t ws_size,
                              hipStream_t stream) {
    const float* hidden = (const float*)d_in[0];
    const float* enc    = (const float*)d_in[1];
    const float* mA     = (const float*)d_in[2];
    const float* mB     = (const float*)d_in[3];
    const float* Wq     = (const float*)d_in[4];
    const float* Wk     = (const float*)d_in[5];
    const float* Wv     = (const float*)d_in[6];
    const float* Wo     = (const float*)d_in[7];
    const float* bo     = (const float*)d_in[8];
    const int*   iA     = (const int*)d_in[9];
    const int*   iB     = (const int*)d_in[10];
    float* out = (float*)d_out;

    const size_t MQ = (size_t)BATCH * HW;   // 32768 query rows
    const size_t MK = (size_t)BATCH * TT;   // 616 key rows

    char* ws = (char*)d_ws;
    size_t off = 0;
    auto carve = [&](size_t bytes) -> void* {
        void* p = ws + off;
        off += (bytes + 255) & ~(size_t)255;
        return p;
    };
    h16* hidden16 = (h16*)carve(MQ * CH * 2);      // reused as ctx after Q-GEMM
    h16* enc16    = (h16*)carve(MK * CTXC * 2);
    h16* WqT      = (h16*)carve((size_t)CH * INNER * 2);    // [N=INNER][K=CH]
    h16* WkT      = (h16*)carve((size_t)CTXC * INNER * 2);  // [N=INNER][K=CTXC]
    h16* WvT      = (h16*)carve((size_t)CTXC * INNER * 2);
    h16* WoT      = (h16*)carve((size_t)INNER * CH * 2);    // [N=CH][K=INNER]
    h16* q16      = (h16*)carve(MQ * INNER * 2);
    h16* k16c     = (h16*)carve(MK * INNER * 2);            // compact K proj
    h16* v16c     = (h16*)carve(MK * INNER * 2);            // compact V proj
    h16* kp       = (h16*)carve((size_t)BATCH * NHEAD * TPAD * HDIM * 2);
    h16* vt       = (h16*)carve((size_t)BATCH * NHEAD * HDIM * TPAD * 2);

    // fp32 -> fp16 (activations plain, weights transposed)
    cvt_f32_f16<<<4096, 256, 0, stream>>>(hidden, hidden16, (long long)MQ * CH);
    cvt_f32_f16<<<1024, 256, 0, stream>>>(enc, enc16, (long long)MK * CTXC);
    cvt_transpose_f32_f16<<<dim3(INNER / 32, CH / 32),   256, 0, stream>>>(Wq, WqT, CH,   INNER);
    cvt_transpose_f32_f16<<<dim3(INNER / 32, CTXC / 32), 256, 0, stream>>>(Wk, WkT, CTXC, INNER);
    cvt_transpose_f32_f16<<<dim3(INNER / 32, CTXC / 32), 256, 0, stream>>>(Wv, WvT, CTXC, INNER);
    cvt_transpose_f32_f16<<<dim3(CH / 32, INNER / 32),   256, 0, stream>>>(Wo, WoT, INNER, CH);

    // projections (WMMA); Q/O have M % 128 == 0 -> async pipelined staging
    gemm_wmma<true, false><<<dim3(256, 16), 128, 0, stream>>>(
        hidden16, WqT, q16, nullptr, nullptr, (int)MQ, INNER, CH);
    gemm_wmma<true, true><<<dim3(5, 16), 128, 0, stream>>>(
        enc16, WkT, k16c, nullptr, nullptr, (int)MK, INNER, CTXC);
    gemm_wmma<true, true><<<dim3(5, 16), 128, 0, stream>>>(
        enc16, WvT, v16c, nullptr, nullptr, (int)MK, INNER, CTXC);

    // zero-padded per-head repack (Kp row-major, Vt transposed)
    repack_kv<<<dim3(TPAD, NHEAD, BATCH), 64, 0, stream>>>(k16c, v16c, kp, vt);

    // attention (WMMA + softmax); ctx overwrites hidden16 workspace
    attn_wmma<<<dim3(HW / 16, NHEAD, BATCH), 32, 0, stream>>>(
        q16, kp, vt, mA, mB, iA, iB, hidden16);

    // output projection + bias + residual (fp32 out)
    gemm_wmma<false, false><<<dim3(256, 16), 128, 0, stream>>>(
        hidden16, WoT, out, bo, hidden, (int)MQ, CH, INNER);
}
